// GCN_39599598469120
// MI455X (gfx1250) — matrix operand order repaired and checked by reading, not compiled, not run on hardware
//
#include <hip/hip_runtime.h>
#include <hip/hip_bf16.h>
#include <stddef.h>
#include <stdint.h>

#define GN 512  // hidden width

typedef __attribute__((ext_vector_type(16))) __bf16 v16bf;
typedef __attribute__((ext_vector_type(8)))  float  v8f;
typedef __attribute__((ext_vector_type(4)))  unsigned int uint4v;
typedef __attribute__((ext_vector_type(8)))  int    int8v;
typedef __attribute__((ext_vector_type(4)))  int    int4v;

union AFrag { unsigned int u[8]; v16bf v; };

__device__ __forceinline__ unsigned short f2bf(float f) {
    union { float f; unsigned int u; } c; c.f = f;
    unsigned int r = c.u + 0x7FFFu + ((c.u >> 16) & 1u);   // round-to-nearest-even
    return (unsigned short)(r >> 16);
}

// ---------------------------------------------------------------- utility
__global__ void zero_f32(float* __restrict__ p, int n) {
    int t = blockIdx.x * 256 + threadIdx.x;
    if (t < n) p[t] = 0.0f;
}

// ------------------------------------------------- layer 1 (F_IN = 1) path
__global__ void scatter_scalar(const float* __restrict__ x,
                               const int* __restrict__ src,
                               const int* __restrict__ dst,
                               float* __restrict__ s, int nedges) {
    int e = blockIdx.x * 256 + threadIdx.x;
    if (e < nedges) atomicAdd(&s[dst[e]], x[src[e]]);
}

__global__ void layer1_kernel(const float* __restrict__ s,
                              const float* __restrict__ x,
                              const float* __restrict__ wrel,
                              const float* __restrict__ brel,
                              const float* __restrict__ wroot,
                              float* __restrict__ out, int nnodes) {
    int t = blockIdx.x * 256 + threadIdx.x;
    if (t >= nnodes * GN) return;
    int i = t >> 9, c = t & (GN - 1);
    float v = s[i] * wrel[c] + brel[c] + x[i] * wroot[c];
    out[t] = v > 0.0f ? v : 0.0f;
}

// -------------------------------------------- 512-wide row scatter (agg)
__global__ void scatter_rows(const float* __restrict__ act,
                             const int* __restrict__ src,
                             const int* __restrict__ dst,
                             float* __restrict__ agg, int nedges) {
    int t = blockIdx.x * 256 + threadIdx.x;   // nedges * 128 threads
    int e  = t >> 7;
    int c4 = (t & 127) * 4;
    if (e >= nedges) return;
    int sn = src[e], dn = dst[e];
    float4 v = *(const float4*)&act[(size_t)sn * GN + c4];
    float* p = &agg[(size_t)dn * GN + c4];
    atomicAdd(p + 0, v.x);
    atomicAdd(p + 1, v.y);
    atomicAdd(p + 2, v.z);
    atomicAdd(p + 3, v.w);
}

// --------------------------------------- weight pre-pack: fp32 -> bf16 pairs
// Output layout: Bpk[colTile(8)][kt(32)][n(64)][kk(16)] packed dwords,
// dword = bf16(W[k0+2kk][colTile*64+n]) | bf16(W[k0+2kk+1][...]) << 16
// (first 16 kt come from Wrel, last 16 from Wroot) -> 4KB contiguous per K-step
__global__ void prepack_w(const float* __restrict__ Wrel,
                          const float* __restrict__ Wroot,
                          unsigned int* __restrict__ Bpk) {
    int t = blockIdx.x * 256 + threadIdx.x;   // 8*32*64*16 = 262144
    int kk  = t & 15;
    int n   = (t >> 4) & 63;
    int ktt = (t >> 10) & 31;
    int c   = t >> 15;
    const float* W = (ktt < 16) ? Wrel : Wroot;
    int k0  = (ktt < 16 ? ktt : ktt - 16) * 32;
    int k   = k0 + kk * 2;
    int col = c * 64 + n;
    unsigned int lo = f2bf(W[(size_t)k * GN + col]);
    unsigned int hi = f2bf(W[(size_t)(k + 1) * GN + col]);
    Bpk[t] = lo | (hi << 16);
}

// --------------------------------------------------------- fused WMMA GEMM
// Out[i,:] = act( Agg[i,:]@Wrel + Prev[i,:]@Wroot + bias ) + Prev[i,:]
// Block: 256 thr (8 waves). Tile: 128 rows x 64 cols. K-step 32 (K total 1024).
// B tiles arrive via the Tensor Data Mover (tensor_load_to_lds, TENSORcnt);
// A tiles are staged + fp32->bf16 converted by the vector pipe in parallel.
__global__ __launch_bounds__(256)
void gemm_fused(const float* __restrict__ Agg, const float* __restrict__ Prev,
                const unsigned int* __restrict__ Bpk,
                const float* __restrict__ bias, float* __restrict__ Out,
                int nrows, int relu) {
    // A tile: 128 rows x 32 k, bf16, row stride padded to 36 (72B, dword aligned)
    __shared__ unsigned short Asm[128 * 36];
    // B tile: packed k-pair dwords, layout [n][kk], contiguous (TDM target)
    __shared__ unsigned int Bsm[64 * 16];

    const int tid  = threadIdx.x;
    const int wave = tid >> 5;
    const int lane = tid & 31;
    const int hi   = lane >> 4;   // lane half selects K sub-range
    const int ln   = lane & 15;
    const int rowBase = blockIdx.x * 128;
    const int colBase = blockIdx.y * 64;

    v8f acc[4];
#pragma unroll
    for (int i = 0; i < 4; ++i)
#pragma unroll
        for (int j = 0; j < 8; ++j) acc[i][j] = 0.0f;

    for (int kt = 0; kt < 32; ++kt) {
        const float* Asrc = (kt < 16) ? Agg : Prev;
        const int k0 = (kt < 16 ? kt : kt - 16) * 32;

        __syncthreads();   // previous iteration's fragment reads done

        // ---- issue TDM DMA: 4KB B tile (global, pre-packed bf16) -> LDS
        if (tid == 0) {
            const unsigned int* bsrc =
                Bpk + ((size_t)blockIdx.y * 32 + kt) * 1024;
            unsigned long long ga = (unsigned long long)(uintptr_t)bsrc;
            unsigned int lds = (unsigned int)(uintptr_t)(void*)Bsm;
            // D# group0: count=1 | lds_addr | global_addr | type=2
            uint4v g0 = { 1u, lds, (unsigned int)ga,
                          (unsigned int)((ga >> 32) & 0x01FFFFFFull) | (2u << 30) };
            // D# group1: data_size=4B; 1-D tile: tensor_dim0=1024,
            // tensor_dim1=1, tile_dim0=1024, stride=1024 dwords
            int8v g1 = { (int)0x00020000, (int)(1024u << 16), (int)(1u << 16),
                         (int)(1024u << 16), 0, 1024, 0, 0 };
            int4v gz4 = { 0, 0, 0, 0 };
            int8v gz8 = { 0, 0, 0, 0, 0, 0, 0, 0 };
            __builtin_amdgcn_tensor_load_to_lds(g0, g1, gz4, gz4, gz8, 0);
        }

        // ---- stage A tile: coalesced float4 rows, convert to bf16
        {
            int r0   = tid >> 3;         // 0..31
            int kseg = (tid & 7) * 4;    // 0..28
#pragma unroll
            for (int i = 0; i < 4; ++i) {
                int r  = r0 + i * 32;
                int gr = rowBase + r;
                float4 v = make_float4(0.f, 0.f, 0.f, 0.f);
                if (gr < nrows)
                    v = *(const float4*)&Asrc[(size_t)gr * GN + k0 + kseg];
                // prefetch next K-step's A segment (global_prefetch_b8)
                if (kt + 1 < 32 && gr < nrows) {
                    const float* nsrc = (kt + 1 < 16) ? Agg : Prev;
                    int nk0 = ((kt + 1 < 16) ? kt + 1 : kt - 15) * 32;
                    __builtin_prefetch(&nsrc[(size_t)gr * GN + nk0 + kseg], 0, 1);
                }
                unsigned short* a = &Asm[r * 36 + kseg];
                a[0] = f2bf(v.x); a[1] = f2bf(v.y);
                a[2] = f2bf(v.z); a[3] = f2bf(v.w);
            }
        }
        if (tid == 0) __builtin_amdgcn_s_wait_tensorcnt(0);
        __syncthreads();   // A stores visible + B DMA complete

        // ---- A fragment for this wave's 16 rows (ISA 16-bit A 16x32 layout)
        AFrag af;
        const unsigned int* Arow =
            (const unsigned int*)&Asm[(wave * 16 + ln) * 36];
#pragma unroll
        for (int u = 0; u < 8; ++u) {
            int idx = (u < 4 ? u : u + 4) + hi * 4;  // K pair = 2*idx
            af.u[u] = Arow[idx];
        }
        // ---- 4 N-subtiles: B fragment + WMMA
#pragma unroll
        for (int sub = 0; sub < 4; ++sub) {
            AFrag bfr;
            const unsigned int* Bn = &Bsm[(sub * 16 + ln) * 16];
#pragma unroll
            for (int u = 0; u < 8; ++u) bfr.u[u] = Bn[(u < 4 ? u : u + 4) + hi * 4];
            acc[sub] = __builtin_amdgcn_wmma_f32_16x16x32_bf16(
                false, af.v, false, bfr.v, (short)0, acc[sub], false, false);
        }
    }

    // ---- epilogue: bias (+ReLU) + residual
#pragma unroll
    for (int sub = 0; sub < 4; ++sub) {
        int col = colBase + sub * 16 + ln;
        float b = bias[col];
#pragma unroll
        for (int g = 0; g < 8; ++g) {
            int row = rowBase + wave * 16 + g + hi * 8;
            if (row < nrows) {
                float v = acc[sub][g] + b;
                if (relu) v = v > 0.0f ? v : 0.0f;
                v += Prev[(size_t)row * GN + col];
                Out[(size_t)row * GN + col] = v;
            }
        }
    }
}

// ------------------------------------------------------------------ driver
extern "C" void kernel_launch(void* const* d_in, const int* in_sizes, int n_in,
                              void* d_out, int out_size, void* d_ws, size_t ws_size,
                              hipStream_t stream) {
    const float* x  = (const float*)d_in[0];
    const int*   ei = (const int*)d_in[1];
    const float* wrel1  = (const float*)d_in[2];
    const float* brel1  = (const float*)d_in[3];
    const float* wroot1 = (const float*)d_in[4];
    const float* wrel2  = (const float*)d_in[5];
    const float* brel2  = (const float*)d_in[6];
    const float* wroot2 = (const float*)d_in[7];
    const float* wrel3  = (const float*)d_in[8];
    const float* brel3  = (const float*)d_in[9];
    const float* wroot3 = (const float*)d_in[10];
    const float* wrel4  = (const float*)d_in[11];
    const float* brel4  = (const float*)d_in[12];
    const float* wroot4 = (const float*)d_in[13];

    const int N = in_sizes[0];       // 100000 (F_IN = 1)
    const int E = in_sizes[1] / 2;   // 400000
    const int* src = ei;
    const int* dst = ei + E;

    // ws layout: s[N] | bufA | bufB | agg (N*512 f32 each) | 3x packed weights
    const size_t BPK_ELEMS = 8u * 32u * 64u * 16u;   // 262144 dwords = 1MB
    float* s    = (float*)d_ws;
    float* bufA = s + (((size_t)N + 255) & ~(size_t)255);
    float* bufB = bufA + (size_t)N * GN;
    float* agg  = bufB + (size_t)N * GN;
    unsigned int* bpk2 = (unsigned int*)(agg + (size_t)N * GN);
    unsigned int* bpk3 = bpk2 + BPK_ELEMS;
    unsigned int* bpk4 = bpk3 + BPK_ELEMS;

    const int total  = N * GN;
    const int gElem  = (total + 255) / 256;
    const int gNodes = (N + 255) / 256;
    const int gEdges = (E + 255) / 256;
    const int gEdgeR = (E * 128 + 255) / 256;
    const int gPack  = (int)(BPK_ELEMS / 256);
    dim3 gGemm((N + 127) / 128, GN / 64);

    // ---- weight pre-pack (bf16, TDM-friendly layout)
    prepack_w<<<gPack, 256, 0, stream>>>(wrel2, wroot2, bpk2);
    prepack_w<<<gPack, 256, 0, stream>>>(wrel3, wroot3, bpk3);
    prepack_w<<<gPack, 256, 0, stream>>>(wrel4, wroot4, bpk4);

    // ---- layer 1 (scalar features)
    zero_f32<<<gNodes, 256, 0, stream>>>(s, N);
    scatter_scalar<<<gEdges, 256, 0, stream>>>(x, src, dst, s, E);
    layer1_kernel<<<gElem, 256, 0, stream>>>(s, x, wrel1, brel1, wroot1, bufA, N);

    // ---- layer 2: relu(conv2) + x1
    zero_f32<<<gElem, 256, 0, stream>>>(agg, total);
    scatter_rows<<<gEdgeR, 256, 0, stream>>>(bufA, src, dst, agg, E);
    gemm_fused<<<gGemm, 256, 0, stream>>>(agg, bufA, bpk2, brel2, bufB, N, 1);

    // ---- layer 3: conv3 + x2
    zero_f32<<<gElem, 256, 0, stream>>>(agg, total);
    scatter_rows<<<gEdgeR, 256, 0, stream>>>(bufB, src, dst, agg, E);
    gemm_fused<<<gGemm, 256, 0, stream>>>(agg, bufB, bpk3, brel3, bufA, N, 0);

    // ---- layer 4: conv4 + x3 -> d_out
    zero_f32<<<gElem, 256, 0, stream>>>(agg, total);
    scatter_rows<<<gEdgeR, 256, 0, stream>>>(bufA, src, dst, agg, E);
    gemm_fused<<<gGemm, 256, 0, stream>>>(agg, bufA, bpk4, brel4,
                                          (float*)d_out, N, 0);
}
